// WaveletChannelAttention_53291954209158
// MI455X (gfx1250) — compile-verified, matrix operand order checked by reading, not run on hardware
//
#include <hip/hip_runtime.h>

// ---------------- types ----------------
typedef __attribute__((ext_vector_type(16))) __bf16    v16bf;
typedef __attribute__((ext_vector_type(8)))  float     v8f;
typedef __attribute__((ext_vector_type(8)))  unsigned  v8u;

// Problem constants: x (2,64,64,64,64) fp32 -> out (2,64,32,32,32) fp32
#define NB    2
#define NC    64
#define CC    512          // cat channels: 64 low + 448 high
#define NO    64           // fuse output channels
#define P     32768        // 32*32*32 output positions per (b,)
#define XCH   262144       // 64*64*64
#define XB    16777216     // 64*XCH
#define S_HAAR 0.35355339059327373f   // (1/sqrt2)^3

#define ROWB  1036         // LDS bytes per position row: 512*2 + 12 pad (lane stride = 3 dwords mod 64 banks)
#define TILE_M 128
#define LDS_BYTES (TILE_M * ROWB)     // 132608

// ws layout (bytes)
#define WS_SUMS   0        // 1024 floats
#define WS_BEFF   4096     // 64 floats
#define WS_WFRAG  8192     // 2*16*4*32 records * 8 dwords = 131072 B

__device__ __forceinline__ unsigned short f2bf(float f) {
  unsigned u = __float_as_uint(f);
  u += 0x7FFFu + ((u >> 16) & 1u);
  return (unsigned short)(u >> 16);
}

// 2x2x2 Haar butterflies. v[idx], idx = i*4 + j*2 + k (i=D, j=H, k=W).
// coef[idx8], idx8 = p*4 + q*2 + r  (0 = low 'aaa'; 1..7 = aad,ada,add,daa,dad,dda,ddd)
__device__ __forceinline__ void haar8(const float v[8], float coef[8]) {
  float a00 = v[0] + v[1], d00 = v[0] - v[1];
  float a01 = v[2] + v[3], d01 = v[2] - v[3];
  float a10 = v[4] + v[5], d10 = v[4] - v[5];
  float a11 = v[6] + v[7], d11 = v[6] - v[7];
  float aa0 = a00 + a01, da0 = a00 - a01;
  float ad0 = d00 + d01, dd0 = d00 - d01;
  float aa1 = a10 + a11, da1 = a10 - a11;
  float ad1 = d10 + d11, dd1 = d10 - d11;
  coef[0] = S_HAAR * (aa0 + aa1);
  coef[1] = S_HAAR * (ad0 + ad1);
  coef[2] = S_HAAR * (da0 + da1);
  coef[3] = S_HAAR * (dd0 + dd1);
  coef[4] = S_HAAR * (aa0 - aa1);
  coef[5] = S_HAAR * (ad0 - ad1);
  coef[6] = S_HAAR * (da0 - da1);
  coef[7] = S_HAAR * (dd0 - dd1);
}

__device__ __forceinline__ void load_block(const float* __restrict__ p, float v[8]) {
  float2 r0 = *reinterpret_cast<const float2*>(p);
  float2 r1 = *reinterpret_cast<const float2*>(p + 64);
  float2 r2 = *reinterpret_cast<const float2*>(p + 4096);
  float2 r3 = *reinterpret_cast<const float2*>(p + 4160);
  v[0] = r0.x; v[1] = r0.y; v[2] = r1.x; v[3] = r1.y;
  v[4] = r2.x; v[5] = r2.y; v[6] = r3.x; v[7] = r3.y;
}

// ---------------- pass 1: DWT + per-cat-channel sums ----------------
// grid = NB*NC*8 blocks of 256; block = (b,c,chunk of 4096 positions), 16 pos/thread
__global__ void __launch_bounds__(256)
k_dwt_sums(const float* __restrict__ x, float* __restrict__ sums) {
  int blk   = blockIdx.x;
  int chunk = blk & 7;
  int c     = (blk >> 3) & 63;
  int b     = blk >> 9;
  const float* xb = x + (size_t)(b * NC + c) * XCH;
  int t = threadIdx.x;

  float acc[8];
#pragma unroll
  for (int s = 0; s < 8; ++s) acc[s] = 0.f;

  for (int i = 0; i < 16; ++i) {
    int p = chunk * 4096 + i * 256 + t;
    int w = p & 31, h = (p >> 5) & 31, d = p >> 10;
    const float* ptr = xb + (2 * d) * 4096 + (2 * h) * 64 + 2 * w;
    float v[8], coef[8];
    load_block(ptr, v);
    haar8(v, coef);
#pragma unroll
    for (int s = 0; s < 8; ++s) acc[s] += coef[s];
  }
  // wave32 reduction
#pragma unroll
  for (int s = 0; s < 8; ++s) {
    float a = acc[s];
    for (int off = 16; off > 0; off >>= 1) a += __shfl_xor(a, off, 32);
    acc[s] = a;
  }
  if ((t & 31) == 0) {
    float* sb = sums + b * CC;
    atomicAdd(&sb[c], acc[0]);
#pragma unroll
    for (int s = 1; s < 8; ++s) atomicAdd(&sb[64 + c * 7 + (s - 1)], acc[s]);
  }
}

// ---------------- prep: means -> MLPs -> att; fold att+BN into bf16 B fragments ----------------
__global__ void __launch_bounds__(256)
k_prep(const float* __restrict__ sums,
       const float* __restrict__ lw1, const float* __restrict__ lw2,
       const float* __restrict__ hw1, const float* __restrict__ hw2,
       const float* __restrict__ fw,  const float* __restrict__ fb,
       const float* __restrict__ g,   const float* __restrict__ be,
       const float* __restrict__ mu,  const float* __restrict__ var,
       float* __restrict__ b_eff, unsigned* __restrict__ w_frag) {
  __shared__ float mean_s[CC];
  __shared__ float att_s[CC];
  __shared__ float hid_s[28];
  int t = threadIdx.x;

  for (int b = 0; b < NB; ++b) {
    for (int i = t; i < CC; i += 256) mean_s[i] = sums[b * CC + i] * (1.f / 32768.f);
    __syncthreads();
    if (t < 4) {
      float s = 0.f;
      for (int c = 0; c < 64; ++c) s += lw1[t * 64 + c] * mean_s[c];
      hid_s[t] = fmaxf(s, 0.f);
    }
    __syncthreads();
    if (t < 64) {
      float s = 0.f;
#pragma unroll
      for (int i = 0; i < 4; ++i) s += lw2[t * 4 + i] * hid_s[i];
      att_s[t] = 1.f / (1.f + expf(-s));
    }
    __syncthreads();
    if (t < 28) {
      float s = 0.f;
      for (int c = 0; c < 448; ++c) s += hw1[t * 448 + c] * mean_s[64 + c];
      hid_s[t] = fmaxf(s, 0.f);
    }
    __syncthreads();
    for (int cc = t; cc < 448; cc += 256) {
      float s = 0.f;
#pragma unroll
      for (int i = 0; i < 28; ++i) s += hw2[cc * 28 + i] * hid_s[i];
      att_s[64 + cc] = 1.f / (1.f + expf(-s));
    }
    __syncthreads();
    // Build bf16 B fragments: rec = kt*128 + j*32 + lane; lane = n + 16*((k>>3)&1);
    // vgpr v holds K pair: k = (v>=4)*16 + (lane>>4)*8 + (v&3)*2
    for (int rec = t; rec < 16 * 4 * 32; rec += 256) {
      int lane = rec & 31, j = (rec >> 5) & 3, kt = rec >> 7;
      int n = lane & 15, g16 = lane >> 4;
      int o = j * 16 + n;
      float scale = g[o] * rsqrtf(var[o] + 1e-5f);
      unsigned base = (unsigned)(b * 2048 + rec) * 8u;
#pragma unroll
      for (int v = 0; v < 8; ++v) {
        int k  = ((v >= 4) ? 16 : 0) + g16 * 8 + (v & 3) * 2;
        int cc = kt * 32 + k;
        float w0 = fw[o * CC + cc]     * att_s[cc]     * scale;
        float w1 = fw[o * CC + cc + 1] * att_s[cc + 1] * scale;
        w_frag[base + v] = (unsigned)f2bf(w0) | ((unsigned)f2bf(w1) << 16);
      }
    }
    __syncthreads();
  }
  if (t < 64) {
    float scale = g[t] * rsqrtf(var[t] + 1e-5f);
    b_eff[t] = fb[t] * scale + be[t] - mu[t] * scale;
  }
}

// ---------------- pass 2: DWT -> LDS (bf16 cat matrix) -> WMMA GEMM -> BN/ReLU store ----------------
// grid = NB*256 blocks of 256 threads; block = 128 output positions, N=64, K=512
__global__ void __launch_bounds__(256)
k_fuse(const float* __restrict__ x, const unsigned* __restrict__ w_frag,
       const float* __restrict__ b_eff, float* __restrict__ out) {
  extern __shared__ unsigned char smem[];
  int blk  = blockIdx.x;
  int tile = blk & 255;
  int b    = blk >> 8;
  int base = tile * TILE_M;                 // flat output position base
  int t    = threadIdx.x;
  const float* xb = x + (size_t)b * XB;

  int d_out = base >> 10;
  int h0    = (base >> 5) & 31;

  // ---- Stage A: cooperative DWT of 64 channels x 128 positions into LDS bf16 ----
  // tau = it*256 + t  =>  m = t & 127 (loop-invariant), c = (t>>7) + 2*it
  {
    int m  = t & 127;
    int c0 = t >> 7;                        // 0 or 1
    int w = m & 31, hh = m >> 5;
    int h = h0 + hh;
    const float* p = xb + (size_t)c0 * XCH + (2 * d_out) * 4096 + (2 * h) * 64 + 2 * w;
    unsigned short* row   = (unsigned short*)(smem + (size_t)m * ROWB);
    unsigned short* lowp  = row + c0;
    unsigned short* highp = row + 64 + c0 * 7;
    for (int it = 0; it < 32; ++it) {
      float v[8], coef[8];
      load_block(p, v);
      haar8(v, coef);
      lowp[0] = f2bf(coef[0]);
#pragma unroll
      for (int s = 1; s < 8; ++s) highp[s - 1] = f2bf(coef[s]);
      p     += 2 * XCH;
      lowp  += 2;
      highp += 14;
    }
  }
  __syncthreads();

  // ---- Stage B: per-wave 16x64x512 GEMM with v_wmma_f32_16x16x32_bf16 ----
  int wave = t >> 5, lane = t & 31;
  int g16  = lane >> 4;
  int mrow = wave * 16 + (lane & 15);
  const unsigned* arow = (const unsigned*)(smem + (size_t)mrow * ROWB + g16 * 16);
  // Per-lane B base; all (kt,j,v) offsets are compile-time constants (<= 64KB)
  const unsigned* bp = w_frag + (size_t)b * 16384 + (unsigned)lane * 8;

  v8f acc[4];
#pragma unroll
  for (int j = 0; j < 4; ++j)
    acc[j] = (v8f){0.f, 0.f, 0.f, 0.f, 0.f, 0.f, 0.f, 0.f};

#pragma unroll
  for (int kt = 0; kt < 16; ++kt) {
    const unsigned* ap = arow + kt * 16;    // kt*64 bytes
    v8u au;
#pragma unroll
    for (int v = 0; v < 4; ++v) au[v] = ap[v];
#pragma unroll
    for (int v = 4; v < 8; ++v) au[v] = ap[4 + v];   // +32B for K>=16 half
    v16bf afrag = __builtin_bit_cast(v16bf, au);

#pragma unroll
    for (int j = 0; j < 4; ++j) {
      v8u bu;
#pragma unroll
      for (int v = 0; v < 8; ++v) bu[v] = bp[kt * 1024 + j * 256 + v];
      v16bf bfrag = __builtin_bit_cast(v16bf, bu);
      acc[j] = __builtin_amdgcn_wmma_f32_16x16x32_bf16(
          false, afrag, false, bfrag, (short)0, acc[j], false, false);
    }
  }
  __syncthreads();   // done reading A staging; reuse LDS for output transpose

  // ---- Stage C: transpose via LDS, fused bias + ReLU, coalesced float4 stores ----
  float* os = (float*)smem;                 // [o][m] stride 132 floats
#pragma unroll
  for (int j = 0; j < 4; ++j) {
    int o = j * 16 + (lane & 15);
#pragma unroll
    for (int r = 0; r < 8; ++r) {
      int m = wave * 16 + g16 * 8 + r;
      os[o * 132 + m] = acc[j][r];
    }
  }
  __syncthreads();

  for (int idx = t; idx < NO * 32; idx += 256) {
    int o = idx >> 5, seg = idx & 31;
    float bias = b_eff[o];
    float4 v4 = *reinterpret_cast<const float4*>(os + o * 132 + seg * 4);
    v4.x = fmaxf(v4.x + bias, 0.f);
    v4.y = fmaxf(v4.y + bias, 0.f);
    v4.z = fmaxf(v4.z + bias, 0.f);
    v4.w = fmaxf(v4.w + bias, 0.f);
    float* op = out + (size_t)(b * NO + o) * P + base;
    *reinterpret_cast<float4*>(op + seg * 4) = v4;
  }
}

// ---------------- launch ----------------
extern "C" void kernel_launch(void* const* d_in, const int* in_sizes, int n_in,
                              void* d_out, int out_size, void* d_ws, size_t ws_size,
                              hipStream_t stream) {
  const float* x   = (const float*)d_in[0];
  const float* lw1 = (const float*)d_in[1];
  const float* lw2 = (const float*)d_in[2];
  const float* hw1 = (const float*)d_in[3];
  const float* hw2 = (const float*)d_in[4];
  const float* fw  = (const float*)d_in[5];
  const float* fb  = (const float*)d_in[6];
  const float* g   = (const float*)d_in[7];
  const float* be  = (const float*)d_in[8];
  const float* mu  = (const float*)d_in[9];
  const float* var = (const float*)d_in[10];

  float*    sums   = (float*)((char*)d_ws + WS_SUMS);
  float*    b_eff  = (float*)((char*)d_ws + WS_BEFF);
  unsigned* w_frag = (unsigned*)((char*)d_ws + WS_WFRAG);

  hipMemsetAsync(sums, 0, NB * CC * sizeof(float), stream);

  k_dwt_sums<<<NB * NC * 8, 256, 0, stream>>>(x, sums);
  k_prep<<<1, 256, 0, stream>>>(sums, lw1, lw2, hw1, hw2, fw, fb, g, be, mu, var,
                                b_eff, w_frag);

  hipFuncSetAttribute(reinterpret_cast<const void*>(k_fuse),
                      hipFuncAttributeMaxDynamicSharedMemorySize, LDS_BYTES);
  k_fuse<<<NB * 256, 256, LDS_BYTES, stream>>>(x, w_frag, b_eff, (float*)d_out);
}
